// PNA_model_26843545600699
// MI455X (gfx1250) — matrix operand order repaired and checked by reading, not compiled, not run on hardware
//
#include <hip/hip_runtime.h>
#include <hip/hip_bf16.h>

typedef __attribute__((ext_vector_type(16))) _Float16 v16h;
typedef __attribute__((ext_vector_type(8)))  _Float16 v8h;
typedef __attribute__((ext_vector_type(8)))  float    v8f;

#define NN   10000     // nodes
#define NE   160000    // edges
#define NG   64        // graphs
#define NT   5         // towers
#define NF   75        // feature dim
#define FPD  80        // padded feature dim (multiple of 16)
#define KE   160       // padded concat dim (2*80)
#define KC_E 5         // K-chunks of 32 in edge GEMM
#define CT_E 5         // col tiles of 16 in edge GEMM (80)
#define FO   15        // tower output dim
#define PIN  975       // post-MLP input dim
#define PINP 992       // padded post-MLP input
#define KC_P 31        // K-chunks of 32 in post GEMM
#define NL   4
#define AVG_LOG 2.8332133f   // log(17.0)
#define BN_EPS  1e-5f
#define FMAXV   3.402823466e38f

// fragment sizes (halves): 32 lanes x 16 halves
#define FRAG_H   512
#define PREW_TOW (KC_E * CT_E * FRAG_H)   // 12800 halves / tower
#define POSTW_TOW (KC_P * FRAG_H)         // 15872 halves / tower

// ---------- float min/max atomics via ordered-int trick ----------
__device__ inline void atomicMinF(float* a, float v) {
    if (v >= 0.f) atomicMin((int*)a, __float_as_int(v));
    else          atomicMax((unsigned int*)a, __float_as_uint(v));
}
__device__ inline void atomicMaxF(float* a, float v) {
    if (v >= 0.f) atomicMax((int*)a, __float_as_int(v));
    else          atomicMin((unsigned int*)a, __float_as_uint(v));
}

// ---------- pre_lin: h = x @ W(2x75) + b ; also f16 mirror ----------
__global__ __launch_bounds__(256) void k_pre_lin(
    const float* __restrict__ x, const float* __restrict__ W,
    const float* __restrict__ b, float* __restrict__ h, _Float16* __restrict__ hh)
{
    int idx = blockIdx.x * 256 + threadIdx.x;
    if (idx >= NN * FPD) return;
    int n = idx / FPD, f = idx % FPD;
    float v = 0.f;
    if (f < NF) v = x[n*2] * W[f] + x[n*2+1] * W[NF + f] + b[f];
    h[idx] = v;
    hh[idx] = (_Float16)v;
}

// ---------- weight conversion into WMMA-fragment-swizzled f16 ----------
// layout: [tower][frag][lane 0..31][i 0..15] where for B operand:
//   K = kc*32 + (lane<16 ? 0 : 8) + (i<8 ? i : i+8),  N = ct*16 + (lane & 15)
__global__ __launch_bounds__(256) void k_cvt_preW(
    const float* __restrict__ preW, _Float16* __restrict__ preWh)
{
    int idx = blockIdx.x * 256 + threadIdx.x;
    if (idx >= NT * PREW_TOW) return;
    int t    = idx / PREW_TOW;
    int r    = idx % PREW_TOW;
    int frag = r / FRAG_H;
    int kc   = frag / CT_E, ct = frag % CT_E;
    int lr   = r % FRAG_H;
    int lane = lr / 16, i = lr % 16;
    int k    = kc * 32 + (lane < 16 ? 0 : 8) + (i < 8 ? i : i + 8);
    int col  = ct * 16 + (lane & 15);
    // padded concat row k -> real concat row c (dst block 0..79, src block 80..159)
    int  c  = (k < FPD) ? k : (NF + (k - FPD));
    bool kv = (k < FPD) ? (k < NF) : ((k - FPD) < NF);
    float v = (kv && col < NF) ? preW[((size_t)t * (2*NF) + c) * NF + col] : 0.f;
    preWh[idx] = (_Float16)v;
}

__global__ __launch_bounds__(256) void k_cvt_postW(
    const float* __restrict__ postW, _Float16* __restrict__ postWh)
{
    int idx = blockIdx.x * 256 + threadIdx.x;
    if (idx >= NT * POSTW_TOW) return;
    int t    = idx / POSTW_TOW;
    int r    = idx % POSTW_TOW;
    int kc   = r / FRAG_H;
    int lr   = r % FRAG_H;
    int lane = lr / 16, i = lr % 16;
    int k    = kc * 32 + (lane < 16 ? 0 : 8) + (i < 8 ? i : i + 8);
    int j    = lane & 15;
    float v  = (k < PIN && j < FO) ? postW[((size_t)t * PIN + k) * FO + j] : 0.f;
    postWh[idx] = (_Float16)v;
}

// ---------- aggregation buffer init ----------
__global__ __launch_bounds__(256) void k_init_agg(
    float* __restrict__ sumB, float* __restrict__ sqB,
    float* __restrict__ mnB,  float* __restrict__ mxB, float* __restrict__ cnt)
{
    size_t idx = (size_t)blockIdx.x * 256 + threadIdx.x;
    if (idx < (size_t)NN * NT * FPD) {
        sumB[idx] = 0.f; sqB[idx] = 0.f; mnB[idx] = FMAXV; mxB[idx] = -FMAXV;
    }
    if (idx < NN) cnt[idx] = 0.f;
}

__global__ __launch_bounds__(256) void k_count(
    const int* __restrict__ eidx, float* __restrict__ cnt)
{
    int e = blockIdx.x * 256 + threadIdx.x;
    if (e < NE) atomicAdd(&cnt[eidx[NE + e]], 1.f);
}

// ---------- edge message GEMM (WMMA f16->f32) + fused multi-aggregate ----------
// grid (ceil(NE/128), NT), block 256 = 8 waves; 16 edges per wave
__global__ __launch_bounds__(256) void k_edge_wmma(
    const _Float16* __restrict__ hh, const _Float16* __restrict__ preWh,
    const float* __restrict__ preb, const int* __restrict__ eidx,
    float* __restrict__ sumB, float* __restrict__ sqB,
    float* __restrict__ mnB,  float* __restrict__ mxB)
{
    __shared__ __align__(16) _Float16 sW[PREW_TOW];  // 25.6KB, fragment-swizzled
    __shared__ int sDst[8][16];
    const int t = blockIdx.y;
    {   // 128-bit LDS staging
        const int4* g = (const int4*)(preWh + (size_t)t * PREW_TOW);
        int4* s = (int4*)sW;
        for (int i = threadIdx.x; i < PREW_TOW * 2 / 16; i += 256) s[i] = g[i];
    }
    const int  w    = threadIdx.x >> 5;
    const int  lane = threadIdx.x & 31;
    const int  hl   = lane & 15;
    const bool hi   = lane >= 16;
    const long ebase = ((long)blockIdx.x * 8 + w) * 16;
    const long rowE  = ebase + hl;
    const bool rv    = rowE < NE;
    const int  srcN  = rv ? eidx[rowE]      : 0;   // edge_index[0] = src
    const int  dstN  = rv ? eidx[NE + rowE] : 0;   // edge_index[1] = dst
    if (lane < 16) sDst[w][hl] = rv ? dstN : -1;
    __syncthreads();

    __builtin_prefetch(hh + (size_t)dstN * FPD, 0, 3);
    __builtin_prefetch(hh + (size_t)srcN * FPD, 0, 3);

    // A fragments: 16 edges x 160 concat features; each 8-half group is one
    // contiguous 16B region of either the dst row or the src row.
    const int klo = hi ? 8 : 0;
    const _Float16* dRow = hh + (size_t)dstN * FPD;
    const _Float16* sRow = hh + (size_t)srcN * FPD;
    v16h a[KC_E];
    #pragma unroll
    for (int kc = 0; kc < KC_E; ++kc) {
        int k0 = kc * 32 + klo;
        int k1 = k0 + 16;
        const _Float16* p0 = (k0 < FPD) ? dRow + k0 : sRow + (k0 - FPD);
        const _Float16* p1 = (k1 < FPD) ? dRow + k1 : sRow + (k1 - FPD);
        v8h lo = *(const v8h*)p0;
        v8h hi8 = *(const v8h*)p1;
        a[kc] = __builtin_shufflevector(lo, hi8, 0,1,2,3,4,5,6,7,8,9,10,11,12,13,14,15);
    }

    #pragma unroll
    for (int ct = 0; ct < CT_E; ++ct) {
        v8f acc = {};
        #pragma unroll
        for (int kc = 0; kc < KC_E; ++kc) {
            v16h b = *(const v16h*)&sW[(kc * CT_E + ct) * FRAG_H + lane * 16];
            acc = __builtin_amdgcn_wmma_f32_16x16x32_f16(
                      false, a[kc], false, b, (short)0, acc, false, false);
        }
        const int col = ct * 16 + hl;
        if (col < NF) {
            const float bias = preb[t * NF + col];
            #pragma unroll
            for (int r = 0; r < 8; ++r) {
                const int M    = r + (hi ? 8 : 0);
                const int node = sDst[w][M];
                if (node >= 0) {
                    const float  v   = acc[r] + bias;
                    const size_t idx = ((size_t)node * NT + t) * FPD + col;
                    atomicAdd(&sumB[idx], v);
                    atomicAdd(&sqB[idx],  v * v);
                    atomicMinF(&mnB[idx], v);
                    atomicMaxF(&mxB[idx], v);
                }
            }
        }
    }
}

// ---------- on-the-fly PNA feature value (975-dim vector element) ----------
__device__ inline float featval(int k, int row, size_t abase,
    const float* __restrict__ h, const float* __restrict__ sumB,
    const float* __restrict__ sqB, const float* __restrict__ mnB,
    const float* __restrict__ mxB, float invd, float amp, float att, bool empty)
{
    if (k >= PIN) return 0.f;
    if (k < NF)   return h[(size_t)row * FPD + k];
    int aI = k - NF;
    int s  = aI / 300;        // scaler: 0=identity, 1=amp, 2=att
    int r  = aI % 300;
    int bl = r / NF;          // 0=mean 1=min 2=max 3=std
    int f  = r % NF;
    float v;
    if (bl == 0)      v = sumB[abase + f] * invd;
    else if (bl == 1) v = empty ? 0.f : mnB[abase + f];
    else if (bl == 2) v = empty ? 0.f : mxB[abase + f];
    else {
        float mean = sumB[abase + f] * invd;
        float ms   = sqB[abase + f] * invd - mean * mean;
        v = sqrtf(fmaxf(ms, 0.f) + 1e-5f);
    }
    if (s == 1) v *= amp;
    else if (s == 2) v *= att;
    return v;
}

// ---------- post-MLP GEMM (WMMA, K=992) with fused feature synthesis ----------
// grid (ceil(NN/128), NT), block 256
__global__ __launch_bounds__(256) void k_post_wmma(
    const float* __restrict__ h, const float* __restrict__ sumB,
    const float* __restrict__ sqB, const float* __restrict__ mnB,
    const float* __restrict__ mxB, const float* __restrict__ cnt,
    const _Float16* __restrict__ postWh, const float* __restrict__ postb,
    float* __restrict__ outbuf)
{
    __shared__ __align__(16) _Float16 sW[POSTW_TOW];  // 31.7KB, fragment-swizzled
    const int t = blockIdx.y;
    {   // 128-bit LDS staging
        const int4* g = (const int4*)(postWh + (size_t)t * POSTW_TOW);
        int4* s = (int4*)sW;
        for (int i = threadIdx.x; i < POSTW_TOW * 2 / 16; i += 256) s[i] = g[i];
    }
    __syncthreads();

    const int  w    = threadIdx.x >> 5;
    const int  lane = threadIdx.x & 31;
    const int  hl   = lane & 15;
    const bool hi   = lane >= 16;
    const int  nb   = (blockIdx.x * 8 + w) * 16;
    const int  row  = min(nb + hl, NN - 1);

    const float  c     = cnt[row];
    const float  d     = fmaxf(c, 1.f);
    const float  invd  = 1.f / d;
    const float  logd  = logf(d + 1.f);
    const float  amp   = logd / AVG_LOG;
    const float  att   = AVG_LOG / logd;
    const bool   empty = (c == 0.f);
    const size_t abase = ((size_t)row * NT + t) * FPD;

    const int klo = hi ? 8 : 0;
    v8f acc = {};
    for (int kc = 0; kc < KC_P; ++kc) {
        v16h a;
        #pragma unroll
        for (int i = 0; i < 8; ++i) {
            a[i]     = (_Float16)featval(kc*32 + klo + i,      row, abase, h, sumB, sqB, mnB, mxB, invd, amp, att, empty);
            a[i + 8] = (_Float16)featval(kc*32 + klo + i + 16, row, abase, h, sumB, sqB, mnB, mxB, invd, amp, att, empty);
        }
        v16h b = *(const v16h*)&sW[kc * FRAG_H + lane * 16];
        acc = __builtin_amdgcn_wmma_f32_16x16x32_f16(
                  false, a, false, b, (short)0, acc, false, false);
    }
    const int j = hl;
    if (j < FO) {
        #pragma unroll
        for (int r = 0; r < 8; ++r) {
            const int M = r + (hi ? 8 : 0);
            const int n = nb + M;
            if (n < NN)
                outbuf[(size_t)n * FPD + t * FO + j] = acc[r] + postb[t * FO + j];
        }
    }
}

// ---------- mixing linear 75x75 ----------
__global__ __launch_bounds__(256) void k_lin(
    const float* __restrict__ outbuf, const float* __restrict__ linW,
    const float* __restrict__ linb, float* __restrict__ outlin)
{
    int idx = blockIdx.x * 256 + threadIdx.x;
    if (idx >= NN * FPD) return;
    int n = idx / FPD, f = idx % FPD;
    if (f >= NF) { outlin[idx] = 0.f; return; }
    float s = linb[f];
    for (int c = 0; c < NF; ++c)
        s += outbuf[(size_t)n * FPD + c] * linW[c * NF + f];
    outlin[idx] = s;
}

// ---------- batchnorm stats (biased var) ----------
__global__ __launch_bounds__(256) void k_bn_stats(
    const float* __restrict__ outlin, float* __restrict__ mu, float* __restrict__ var)
{
    __shared__ float s1[256], s2[256];
    int f = blockIdx.x;  // < NF
    float a = 0.f, b = 0.f;
    for (int n = threadIdx.x; n < NN; n += 256) {
        float v = outlin[(size_t)n * FPD + f];
        a += v; b += v * v;
    }
    s1[threadIdx.x] = a; s2[threadIdx.x] = b; __syncthreads();
    for (int s = 128; s > 0; s >>= 1) {
        if (threadIdx.x < s) { s1[threadIdx.x] += s1[threadIdx.x + s];
                               s2[threadIdx.x] += s2[threadIdx.x + s]; }
        __syncthreads();
    }
    if (threadIdx.x == 0) {
        float m = s1[0] / NN;
        mu[f] = m;
        var[f] = s2[0] / NN - m * m;
    }
}

__global__ __launch_bounds__(256) void k_bn_apply(
    const float* __restrict__ outlin, const float* __restrict__ mu,
    const float* __restrict__ var, const float* __restrict__ scale,
    const float* __restrict__ bias, float* __restrict__ h, _Float16* __restrict__ hh)
{
    int idx = blockIdx.x * 256 + threadIdx.x;
    if (idx >= NN * FPD) return;
    int f = idx % FPD;
    float v = 0.f;
    if (f < NF) {
        v = (outlin[idx] - mu[f]) * rsqrtf(var[f] + BN_EPS) * scale[f] + bias[f];
        v = fmaxf(v, 0.f);
    }
    h[idx] = v;
    hh[idx] = (_Float16)v;
}

// ---------- pooling + head ----------
__global__ __launch_bounds__(256) void k_zero(float* __restrict__ p, int n)
{
    int i = blockIdx.x * 256 + threadIdx.x;
    if (i < n) p[i] = 0.f;
}

__global__ __launch_bounds__(256) void k_pool(
    const float* __restrict__ h, const int* __restrict__ batch, float* __restrict__ pooled)
{
    int idx = blockIdx.x * 256 + threadIdx.x;
    if (idx >= NN * FPD) return;
    int n = idx / FPD, f = idx % FPD;
    if (f >= NF) return;
    atomicAdd(&pooled[(size_t)batch[n] * FPD + f], h[idx]);
}

__global__ __launch_bounds__(64) void k_head(
    const float* __restrict__ pooled,
    const float* __restrict__ W1, const float* __restrict__ b1,
    const float* __restrict__ W2, const float* __restrict__ b2,
    const float* __restrict__ W3, const float* __restrict__ b3,
    float* __restrict__ out)
{
    int g = threadIdx.x;
    if (g >= NG) return;
    float z1[50];
    for (int j = 0; j < 50; ++j) {
        float s = b1[j];
        for (int c = 0; c < NF; ++c) s += pooled[(size_t)g * FPD + c] * W1[c * 50 + j];
        z1[j] = fmaxf(s, 0.f);
    }
    float z2[25];
    for (int j = 0; j < 25; ++j) {
        float s = b2[j];
        for (int c = 0; c < 50; ++c) s += z1[c] * W2[c * 25 + j];
        z2[j] = fmaxf(s, 0.f);
    }
    float s = b3[0];
    for (int c = 0; c < 25; ++c) s += z2[c] * W3[c];
    out[g] = s;
}

// ================= host launch =================
extern "C" void kernel_launch(void* const* d_in, const int* in_sizes, int n_in,
                              void* d_out, int out_size, void* d_ws, size_t ws_size,
                              hipStream_t stream)
{
    (void)in_sizes; (void)n_in; (void)out_size; (void)ws_size;
    const float* x        = (const float*)d_in[0];
    const int*   eidx     = (const int*)  d_in[1];
    const int*   batch    = (const int*)  d_in[2];
    const float* preLW    = (const float*)d_in[3];
    const float* preLb    = (const float*)d_in[4];
    const float* pre_W    = (const float*)d_in[5];
    const float* pre_b    = (const float*)d_in[6];
    const float* post_W   = (const float*)d_in[7];
    const float* post_b   = (const float*)d_in[8];
    const float* lin_W    = (const float*)d_in[9];
    const float* lin_b    = (const float*)d_in[10];
    const float* bn_scale = (const float*)d_in[11];
    const float* bn_bias  = (const float*)d_in[12];
    const float* W1 = (const float*)d_in[13]; const float* b1 = (const float*)d_in[14];
    const float* W2 = (const float*)d_in[15]; const float* b2 = (const float*)d_in[16];
    const float* W3 = (const float*)d_in[17]; const float* b3 = (const float*)d_in[18];
    float* outp = (float*)d_out;

    // workspace carve-out (256B aligned)
    char* ws = (char*)d_ws;
    size_t off = 0;
    auto carve = [&](size_t bytes) -> void* {
        void* p = ws + off;
        off = (off + bytes + 255) & ~(size_t)255;
        return p;
    };
    float*    h      = (float*)   carve((size_t)NN * FPD * 4);
    _Float16* hh     = (_Float16*)carve((size_t)NN * FPD * 2);
    float*    sumB   = (float*)   carve((size_t)NN * NT * FPD * 4);
    float*    sqB    = (float*)   carve((size_t)NN * NT * FPD * 4);
    float*    mnB    = (float*)   carve((size_t)NN * NT * FPD * 4);
    float*    mxB    = (float*)   carve((size_t)NN * NT * FPD * 4);
    float*    cnt    = (float*)   carve((size_t)NN * 4);
    float*    outbuf = (float*)   carve((size_t)NN * FPD * 4);
    float*    outlin = (float*)   carve((size_t)NN * FPD * 4);
    float*    mu     = (float*)   carve(FPD * 4);
    float*    var    = (float*)   carve(FPD * 4);
    float*    pooled = (float*)   carve((size_t)NG * FPD * 4);
    _Float16* preWh  = (_Float16*)carve((size_t)NT * PREW_TOW * 2);
    _Float16* postWh = (_Float16*)carve((size_t)NT * POSTW_TOW * 2);

    const int G_NF = (NN * FPD + 255) / 256;          // 3125
    k_pre_lin<<<G_NF, 256, 0, stream>>>(x, preLW, preLb, h, hh);

    for (int l = 0; l < NL; ++l) {
        k_cvt_preW <<<(NT*PREW_TOW + 255)/256,  256, 0, stream>>>(pre_W  + (size_t)l*NT*2*NF*NF, preWh);
        k_cvt_postW<<<(NT*POSTW_TOW + 255)/256, 256, 0, stream>>>(post_W + (size_t)l*NT*PIN*FO,  postWh);
        k_init_agg <<<(NN*NT*FPD + 255)/256,    256, 0, stream>>>(sumB, sqB, mnB, mxB, cnt);
        k_count    <<<(NE + 255)/256,           256, 0, stream>>>(eidx, cnt);
        k_edge_wmma<<<dim3((NE + 127)/128, NT), 256, 0, stream>>>(
            hh, preWh, pre_b + (size_t)l*NT*NF, eidx, sumB, sqB, mnB, mxB);
        k_post_wmma<<<dim3((NN + 127)/128, NT), 256, 0, stream>>>(
            h, sumB, sqB, mnB, mxB, cnt, postWh, post_b + (size_t)l*NT*FO, outbuf);
        k_lin      <<<G_NF, 256, 0, stream>>>(outbuf, lin_W + (size_t)l*NF*NF, lin_b + (size_t)l*NF, outlin);
        k_bn_stats <<<NF,   256, 0, stream>>>(outlin, mu, var);
        k_bn_apply <<<G_NF, 256, 0, stream>>>(outlin, mu, var,
            bn_scale + (size_t)l*NF, bn_bias + (size_t)l*NF, h, hh);
    }

    k_zero<<<(NG*FPD + 255)/256, 256, 0, stream>>>(pooled, NG * FPD);
    k_pool<<<G_NF, 256, 0, stream>>>(h, batch, pooled);
    k_head<<<1, 64, 0, stream>>>(pooled, W1, b1, W2, b2, W3, b3, outp);
}
